// SGFormer_21028159881637
// MI455X (gfx1250) — matrix operand order repaired and checked by reading, not compiled, not run on hardware
//
#include <hip/hip_runtime.h>
#include <hip/hip_bf16.h>
#include <math.h>

#define N_NODES 100000
#define N_EDGES 800000
#define HID 128
#define OUT_F 64
#define LN_EPS 1e-5f

typedef __attribute__((ext_vector_type(16))) _Float16 v16h;
typedef __attribute__((ext_vector_type(8)))  float    v8f;

// ---------------- WMMA fragment helpers ----------------
// A fragment: 16x32 f16, built from an f32 row-major source (scaled, f32->f16).
// Layout: lanes 0-15 (M=lane): K = {0..15} contiguous;
// lanes 16-31 (M=lane-16): K = {16..31}.
static __device__ inline v16h a_frag_f32(const float* S, int rowbase, int nrows,
                                         int ld, int k0, float scale) {
  int lane = threadIdx.x & 31;
  int m = lane & 15;
  int kb = (lane & 16) ? 16 : 0;
  int row = rowbase + m;
  v16h a;
  if (row < nrows) {
    const float* p = S + (size_t)row * ld + k0 + kb;
#pragma unroll
    for (int i = 0; i < 8; ++i) {
      a[i]     = (_Float16)(p[i] * scale);
      a[i + 8] = (_Float16)(p[i + 8] * scale);
    }
  } else {
#pragma unroll
    for (int i = 0; i < 16; ++i) a[i] = (_Float16)0.0f;
  }
  return a;
}

// B fragment: 32x16 f16 from a pre-transposed weight Wt[fo][fi] (row-major, ld=fi).
// Lanes 0-15 (N=lane): K = kg+{0..15} contiguous; lanes 16-31: kg=16.
static __device__ inline v16h b_frag_f16(const _Float16* Wt, int n0, int ld, int k0) {
  int lane = threadIdx.x & 31;
  int n = n0 + (lane & 15);
  int kg = (lane & 16) ? 16 : 0;
  return *(const v16h*)(Wt + (size_t)n * ld + k0 + kg);
}

// 16-row strip x 128-col GEMM core: K=128, accumulates into acc[8] (one 16x16 tile per nt).
static __device__ inline void gemm128(const float* A, int rowbase, int nrows,
                                      const _Float16* Wt, float ascale, v8f acc[8]) {
#pragma unroll
  for (int ks = 0; ks < 128; ks += 32) {
    v16h a = a_frag_f32(A, rowbase, nrows, HID, ks, ascale);
#pragma unroll
    for (int nt = 0; nt < 8; ++nt) {
      v16h b = b_frag_f16(Wt, nt * 16, HID, ks);
      acc[nt] = __builtin_amdgcn_wmma_f32_16x16x32_f16(false, a, false, b,
                                                       (short)0, acc[nt], false, false);
    }
  }
}

static __device__ inline void zero_acc8(v8f acc[8]) {
  v8f z = {0.f, 0.f, 0.f, 0.f, 0.f, 0.f, 0.f, 0.f};
#pragma unroll
  for (int i = 0; i < 8; ++i) acc[i] = z;
}

// ---------------- small utility kernels ----------------
__global__ void k_wtrans(const float* __restrict__ W, _Float16* __restrict__ Wt,
                         int R, int C) {
  int i = blockIdx.x * blockDim.x + threadIdx.x;
  if (i < R * C) {
    int r = i / C, c = i % C;
    Wt[(size_t)c * R + r] = (_Float16)W[i];
  }
}

__global__ void k_zero(float* __restrict__ p, int n) {
  int i = blockIdx.x * blockDim.x + threadIdx.x;
  if (i < n) p[i] = 0.0f;
}

// ---------------- GEMM + LayerNorm + ReLU (transformer input proj) ----------------
__global__ void k_gemm_ln_relu(const float* __restrict__ A, const _Float16* __restrict__ Wt,
                               const float* __restrict__ bias, const float* __restrict__ gamma,
                               const float* __restrict__ beta, float* __restrict__ Out) {
  int wave = threadIdx.x >> 5, lane = threadIdx.x & 31;
  int rowbase = blockIdx.x * 128 + wave * 16;
  if (rowbase >= N_NODES) return;
  v8f acc[8];
  zero_acc8(acc);
  gemm128(A, rowbase, N_NODES, Wt, 1.0f, acc);
  int cbase = lane & 15;
  int rhalf = (lane & 16) ? 8 : 0;
#pragma unroll
  for (int r = 0; r < 8; ++r) {
    float s = 0.f, s2 = 0.f;
#pragma unroll
    for (int nt = 0; nt < 8; ++nt) {
      float t = acc[nt][r] + bias[nt * 16 + cbase];
      s += t; s2 += t * t;
    }
#pragma unroll
    for (int m = 1; m <= 8; m <<= 1) {
      s  += __shfl_xor(s, m, 32);
      s2 += __shfl_xor(s2, m, 32);
    }
    float mean = s * (1.0f / 128.0f);
    float var  = s2 * (1.0f / 128.0f) - mean * mean;
    float rstd = rsqrtf(var + LN_EPS);
    int row = rowbase + r + rhalf;
    if (row < N_NODES) {
#pragma unroll
      for (int nt = 0; nt < 8; ++nt) {
        int col = nt * 16 + cbase;
        float t = acc[nt][r] + bias[col];
        float o = (t - mean) * rstd * gamma[col] + beta[col];
        Out[(size_t)row * HID + col] = fmaxf(o, 0.0f);
      }
    }
  }
}

// ---------------- GEMM + bias + optional global stats (q/k/v) ----------------
__global__ void k_gemm_bias_stats(const float* __restrict__ A, const _Float16* __restrict__ Wt,
                                  const float* __restrict__ bias, float* __restrict__ Out,
                                  float* sumsq, float* colsum) {
  int wave = threadIdx.x >> 5, lane = threadIdx.x & 31;
  int rowbase = blockIdx.x * 128 + wave * 16;
  if (rowbase >= N_NODES) return;
  v8f acc[8];
  zero_acc8(acc);
  gemm128(A, rowbase, N_NODES, Wt, 1.0f, acc);
  int cbase = lane & 15;
  int rhalf = (lane & 16) ? 8 : 0;
  float lsq = 0.f;
  float colacc[8];
#pragma unroll
  for (int nt = 0; nt < 8; ++nt) colacc[nt] = 0.f;
#pragma unroll
  for (int r = 0; r < 8; ++r) {
    int row = rowbase + r + rhalf;
    if (row < N_NODES) {
#pragma unroll
      for (int nt = 0; nt < 8; ++nt) {
        int col = nt * 16 + cbase;
        float val = acc[nt][r] + bias[col];
        Out[(size_t)row * HID + col] = val;
        lsq += val * val;
        colacc[nt] += val;
      }
    }
  }
  if (sumsq) {
#pragma unroll
    for (int m = 1; m <= 16; m <<= 1) lsq += __shfl_xor(lsq, m, 32);
    if (lane == 0) atomicAdd(sumsq, lsq);
  }
  if (colsum) {
#pragma unroll
    for (int nt = 0; nt < 8; ++nt) {
      float cs = colacc[nt] + __shfl_xor(colacc[nt], 16, 32);
      if (lane < 16) atomicAdd(&colsum[nt * 16 + cbase], cs);
    }
  }
}

// ---------------- kvs = K^T V  (128x128, reduce over N via LDS-staged WMMA) --------
__global__ void k_kvs(const float* __restrict__ K, const float* __restrict__ V,
                      float* __restrict__ kvs) {
  __shared__ _Float16 kT[128 * 34];   // padded stride 34 halves -> conflict-free
  __shared__ _Float16 vT[128 * 34];
  int wave = threadIdx.x >> 5, lane = threadIdx.x & 31;
  v8f acc[8];
  zero_acc8(acc);
  int node8 = threadIdx.x >> 3;       // 0..31  (node within chunk)
  int fb    = (threadIdx.x & 7) * 16; // feature base, 16 feats per thread
  for (int ch = blockIdx.x; ch < N_NODES / 32; ch += gridDim.x) {
    __syncthreads();
    int node0 = ch * 32;
    const float* kp = K + (size_t)(node0 + node8) * HID + fb;
    const float* vp = V + (size_t)(node0 + node8) * HID + fb;
#pragma unroll
    for (int j = 0; j < 16; ++j) {
      kT[(fb + j) * 34 + node8] = (_Float16)kp[j];
      vT[(fb + j) * 34 + node8] = (_Float16)vp[j];
    }
    __syncthreads();
    // A(m,kk) = k[node0+kk, 16*wave+m]   (transposed access out of LDS)
    v16h a;
    int mf = wave * 16 + (lane & 15);
    int kb = (lane & 16) ? 16 : 0;
#pragma unroll
    for (int e = 0; e < 8; ++e) {
      a[e]     = kT[mf * 34 + kb + e];
      a[e + 8] = kT[mf * 34 + kb + 8 + e];
    }
#pragma unroll
    for (int nt = 0; nt < 8; ++nt) {
      int nf = nt * 16 + (lane & 15);
      int kg = (lane & 16) ? 16 : 0;
      v16h b;
#pragma unroll
      for (int e = 0; e < 16; ++e) b[e] = vT[nf * 34 + kg + e];
      acc[nt] = __builtin_amdgcn_wmma_f32_16x16x32_f16(false, a, false, b,
                                                       (short)0, acc[nt], false, false);
    }
  }
  int rhalf = (lane & 16) ? 8 : 0;
#pragma unroll
  for (int nt = 0; nt < 8; ++nt)
#pragma unroll
    for (int r = 0; r < 8; ++r)
      atomicAdd(&kvs[(size_t)(wave * 16 + r + rhalf) * HID + nt * 16 + (lane & 15)],
                acc[nt][r]);
}

// kvsT[d][m] = kvs[m][d] / ||k||   (f16, transposed for B-fragment loads)
__global__ void k_kvs_convert(const float* __restrict__ kvs, const float* __restrict__ scal,
                              _Float16* __restrict__ kvsT) {
  int i = blockIdx.x * blockDim.x + threadIdx.x;
  if (i < HID * HID) {
    int m = i >> 7, d = i & 127;
    float rsk = rsqrtf(scal[1]);
    kvsT[(size_t)d * HID + m] = (_Float16)(kvs[i] * rsk);
  }
}

// denom[i] = (q[i,:]/||q||) . (ksum/||k||) + N
__global__ void k_denomv(const float* __restrict__ Q, const float* __restrict__ ksum,
                         const float* __restrict__ scal, float* __restrict__ denomv) {
  int i = blockIdx.x * blockDim.x + threadIdx.x;
  if (i < N_NODES) {
    float dot = 0.f;
    const float* q = Q + (size_t)i * HID;
#pragma unroll 8
    for (int c = 0; c < HID; ++c) dot += q[c] * ksum[c];
    denomv[i] = dot * rsqrtf(scal[0]) * rsqrtf(scal[1]) + (float)N_NODES;
  }
}

// z = ((q/||q||) @ kvsT + N*v)/denom ; t = 0.5*(z+h) ; LN ; ReLU -> x_trans
__global__ void k_attn_out(const float* __restrict__ Q, const _Float16* __restrict__ kvsT,
                           const float* __restrict__ scal, const float* __restrict__ V,
                           const float* __restrict__ H, const float* __restrict__ denomv,
                           const float* __restrict__ gamma, const float* __restrict__ beta,
                           float* __restrict__ Out) {
  int wave = threadIdx.x >> 5, lane = threadIdx.x & 31;
  int rowbase = blockIdx.x * 128 + wave * 16;
  if (rowbase >= N_NODES) return;
  float rsq = rsqrtf(scal[0]);
  v8f acc[8];
  zero_acc8(acc);
  gemm128(Q, rowbase, N_NODES, kvsT, rsq, acc);
  int cbase = lane & 15;
  int rhalf = (lane & 16) ? 8 : 0;
  // overwrite acc in place with t = 0.5*(z + h)
#pragma unroll
  for (int r = 0; r < 8; ++r) {
    int row = rowbase + r + rhalf;
    float dn = (row < N_NODES) ? denomv[row] : 1.0f;
#pragma unroll
    for (int nt = 0; nt < 8; ++nt) {
      int col = nt * 16 + cbase;
      float vv = (row < N_NODES) ? V[(size_t)row * HID + col] : 0.f;
      float hh = (row < N_NODES) ? H[(size_t)row * HID + col] : 0.f;
      float z = (acc[nt][r] + 100000.0f * vv) / dn;
      acc[nt][r] = 0.5f * (z + hh);
    }
  }
#pragma unroll
  for (int r = 0; r < 8; ++r) {
    float s = 0.f, s2 = 0.f;
#pragma unroll
    for (int nt = 0; nt < 8; ++nt) { float t = acc[nt][r]; s += t; s2 += t * t; }
#pragma unroll
    for (int m = 1; m <= 8; m <<= 1) {
      s  += __shfl_xor(s, m, 32);
      s2 += __shfl_xor(s2, m, 32);
    }
    float mean = s * (1.0f / 128.0f);
    float var  = s2 * (1.0f / 128.0f) - mean * mean;
    float rstd = rsqrtf(var + LN_EPS);
    int row = rowbase + r + rhalf;
    if (row < N_NODES) {
#pragma unroll
      for (int nt = 0; nt < 8; ++nt) {
        int col = nt * 16 + cbase;
        float o = (acc[nt][r] - mean) * rstd * gamma[col] + beta[col];
        Out[(size_t)row * HID + col] = fmaxf(o, 0.0f);
      }
    }
  }
}

// ---------------- GEMM + BatchNorm + ReLU (+ optional residual) ----------------
__global__ void k_gemm_bn_relu(const float* __restrict__ A, const _Float16* __restrict__ Wt,
                               const float* __restrict__ bias,
                               const float* __restrict__ bg, const float* __restrict__ bb,
                               const float* __restrict__ bm, const float* __restrict__ bv,
                               const float* __restrict__ res, float* __restrict__ Out) {
  int wave = threadIdx.x >> 5, lane = threadIdx.x & 31;
  int rowbase = blockIdx.x * 128 + wave * 16;
  if (rowbase >= N_NODES) return;
  v8f acc[8];
  zero_acc8(acc);
  gemm128(A, rowbase, N_NODES, Wt, 1.0f, acc);
  int cbase = lane & 15;
  int rhalf = (lane & 16) ? 8 : 0;
#pragma unroll
  for (int r = 0; r < 8; ++r) {
    int row = rowbase + r + rhalf;
    if (row < N_NODES) {
#pragma unroll
      for (int nt = 0; nt < 8; ++nt) {
        int col = nt * 16 + cbase;
        float t = acc[nt][r] + bias[col];
        float o = (t - bm[col]) * rsqrtf(bv[col] + LN_EPS) * bg[col] + bb[col];
        o = fmaxf(o, 0.0f);
        if (res) o += res[(size_t)row * HID + col];
        Out[(size_t)row * HID + col] = o;
      }
    }
  }
}

// ---------------- graph kernels ----------------
__global__ void k_degree(const int* __restrict__ ei, const int* __restrict__ et,
                         float* __restrict__ deg0, float* __restrict__ deg1) {
  int e = blockIdx.x * blockDim.x + threadIdx.x;
  if (e < N_EDGES) {
    int c = ei[N_EDGES + e];
    float* d = (et[e] == 0) ? deg0 : deg1;
    atomicAdd(&d[c], 1.0f);
  }
}

// agg[col] += ew/sqrt(d[col]*d[row]) * G[row]  (32 lanes per edge, 4 feats/lane)
__global__ void k_spmm(const float* __restrict__ G, float* __restrict__ agg,
                       const int* __restrict__ ei, const int* __restrict__ et,
                       const float* __restrict__ ew, const float* __restrict__ deg,
                       int wtype) {
  int e = blockIdx.x * 8 + (threadIdx.x >> 5);
  int lane = threadIdx.x & 31;
  if (e >= N_EDGES) return;
  if (et[e] != wtype) return;
  int r = ei[e], c = ei[N_EDGES + e];
  float dd = deg[c] * deg[r];
  if (!(dd > 0.0f)) return;              // nan_to_num(inf/nan) -> 0
  float val = ew[e] * rsqrtf(dd);
  int f0 = lane * 4;
  const float4 gv = *(const float4*)(G + (size_t)r * HID + f0);
  float* ap = agg + (size_t)c * HID + f0;
  atomicAdd(ap + 0, val * gv.x);
  atomicAdd(ap + 1, val * gv.y);
  atomicAdd(ap + 2, val * gv.z);
  atomicAdd(ap + 3, val * gv.w);
}

// ---------------- final: tanh(concat[x_trans, g2] @ out_w + b) ----------------
__global__ void k_out(const float* __restrict__ Xt, const float* __restrict__ G2,
                      const _Float16* __restrict__ Wt, const float* __restrict__ bias,
                      float* __restrict__ Out) {
  int wave = threadIdx.x >> 5, lane = threadIdx.x & 31;
  int rowbase = blockIdx.x * 128 + wave * 16;
  if (rowbase >= N_NODES) return;
  v8f acc[4];
  v8f z = {0.f, 0.f, 0.f, 0.f, 0.f, 0.f, 0.f, 0.f};
#pragma unroll
  for (int i = 0; i < 4; ++i) acc[i] = z;
#pragma unroll
  for (int ks = 0; ks < 256; ks += 32) {
    const float* src = (ks < 128) ? Xt : G2;
    int k0 = (ks < 128) ? ks : ks - 128;
    v16h a = a_frag_f32(src, rowbase, N_NODES, HID, k0, 1.0f);
#pragma unroll
    for (int nt = 0; nt < 4; ++nt) {
      v16h b = b_frag_f16(Wt, nt * 16, 256, ks);
      acc[nt] = __builtin_amdgcn_wmma_f32_16x16x32_f16(false, a, false, b,
                                                       (short)0, acc[nt], false, false);
    }
  }
  int cbase = lane & 15;
  int rhalf = (lane & 16) ? 8 : 0;
#pragma unroll
  for (int r = 0; r < 8; ++r) {
    int row = rowbase + r + rhalf;
    if (row < N_NODES) {
#pragma unroll
      for (int nt = 0; nt < 4; ++nt) {
        int col = nt * 16 + cbase;
        Out[(size_t)row * OUT_F + col] = tanhf(acc[nt][r] + bias[col]);
      }
    }
  }
}

// ---------------- host-side launch ----------------
extern "C" void kernel_launch(void* const* d_in, const int* in_sizes, int n_in,
                              void* d_out, int out_size, void* d_ws, size_t ws_size,
                              hipStream_t stream) {
  const float* x       = (const float*)d_in[0];
  const float* ew      = (const float*)d_in[1];
  const float* t_fc_w  = (const float*)d_in[2];
  const float* t_fc_b  = (const float*)d_in[3];
  const float* t_ln0_g = (const float*)d_in[4];
  const float* t_ln0_b = (const float*)d_in[5];
  const float* wq      = (const float*)d_in[6];
  const float* wq_b    = (const float*)d_in[7];
  const float* wk      = (const float*)d_in[8];
  const float* wk_b    = (const float*)d_in[9];
  const float* wv      = (const float*)d_in[10];
  const float* wv_b    = (const float*)d_in[11];
  const float* t_ln1_g = (const float*)d_in[12];
  const float* t_ln1_b = (const float*)d_in[13];
  const float* g_fc_w  = (const float*)d_in[14];
  const float* g_fc_b  = (const float*)d_in[15];
  const float* bn0_g   = (const float*)d_in[16];
  const float* bn0_b   = (const float*)d_in[17];
  const float* bn0_m   = (const float*)d_in[18];
  const float* bn0_v   = (const float*)d_in[19];
  const float* g_w1    = (const float*)d_in[20];
  const float* g_b1    = (const float*)d_in[21];
  const float* bn1_g   = (const float*)d_in[22];
  const float* bn1_b   = (const float*)d_in[23];
  const float* bn1_m   = (const float*)d_in[24];
  const float* bn1_v   = (const float*)d_in[25];
  const float* g_w2    = (const float*)d_in[26];
  const float* g_b2    = (const float*)d_in[27];
  const float* bn2_g   = (const float*)d_in[28];
  const float* bn2_b   = (const float*)d_in[29];
  const float* bn2_m   = (const float*)d_in[30];
  const float* bn2_v   = (const float*)d_in[31];
  const float* out_w   = (const float*)d_in[32];
  const float* out_b   = (const float*)d_in[33];
  const int*   ei      = (const int*)d_in[34];
  const int*   et      = (const int*)d_in[35];

  char* ws = (char*)d_ws;
  const size_t NB = (size_t)N_NODES * HID * sizeof(float);   // 51.2 MB
  float* h   = (float*)(ws + 0 * NB);
  float* q   = (float*)(ws + 1 * NB);
  float* k   = (float*)(ws + 2 * NB);
  float* v   = (float*)(ws + 3 * NB);
  float* xt  = (float*)(ws + 4 * NB);
  float* g   = h;    // reuse after transformer branch completes (stream-ordered)
  float* g1  = q;
  float* g2  = k;
  float* agg = v;
  char* sm = ws + 5 * NB;
  float*    kvs    = (float*)sm;                    //  65536 B
  _Float16* kvsT   = (_Float16*)(sm + 65536);       //  32768 B
  float*    ksum   = (float*)(sm + 98304);          //    512 B (then scal)
  float*    scal   = (float*)(sm + 98816);          // [0]=||q||^2, [1]=||k||^2
  float*    denomv = (float*)(sm + 99072);          // 400000 B
  float*    deg0   = (float*)(sm + 499072);
  float*    deg1   = (float*)(sm + 899072);
  _Float16* wT     = (_Float16*)(sm + 1299072);     // 8 slots * 16384 halves
  _Float16* tfcT = wT + 0 * 16384;
  _Float16* wqT  = wT + 1 * 16384;
  _Float16* wkT  = wT + 2 * 16384;
  _Float16* wvT  = wT + 3 * 16384;
  _Float16* gfcT = wT + 4 * 16384;
  _Float16* gw1T = wT + 5 * 16384;
  _Float16* gw2T = wT + 6 * 16384;
  _Float16* outT = wT + 7 * 16384;

  const int GB = (N_NODES + 127) / 128;  // 782 blocks of 8 waves * 16 rows

  // --- weight convert + transpose (f32 -> f16, [fo][fi]) ---
  k_wtrans<<<64, 256, 0, stream>>>(t_fc_w, tfcT, 128, 128);
  k_wtrans<<<64, 256, 0, stream>>>(wq,     wqT,  128, 128);
  k_wtrans<<<64, 256, 0, stream>>>(wk,     wkT,  128, 128);
  k_wtrans<<<64, 256, 0, stream>>>(wv,     wvT,  128, 128);
  k_wtrans<<<64, 256, 0, stream>>>(g_fc_w, gfcT, 128, 128);
  k_wtrans<<<64, 256, 0, stream>>>(g_w1,   gw1T, 128, 128);
  k_wtrans<<<64, 256, 0, stream>>>(g_w2,   gw2T, 128, 128);
  k_wtrans<<<64, 256, 0, stream>>>(out_w,  outT, 256, 64);

  // --- zero accumulators ---
  k_zero<<<64, 256, 0, stream>>>(kvs, HID * HID);
  k_zero<<<1, 256, 0, stream>>>(ksum, 130);                 // ksum + scal
  k_zero<<<(2 * N_NODES + 255) / 256, 256, 0, stream>>>(deg0, 2 * N_NODES);

  // --- transformer branch ---
  k_gemm_ln_relu<<<GB, 256, 0, stream>>>(x, tfcT, t_fc_b, t_ln0_g, t_ln0_b, h);
  k_gemm_bias_stats<<<GB, 256, 0, stream>>>(h, wqT, wq_b, q, &scal[0], nullptr);
  k_gemm_bias_stats<<<GB, 256, 0, stream>>>(h, wkT, wk_b, k, &scal[1], ksum);
  k_gemm_bias_stats<<<GB, 256, 0, stream>>>(h, wvT, wv_b, v, nullptr, nullptr);
  k_kvs<<<208, 256, 0, stream>>>(k, v, kvs);
  k_kvs_convert<<<64, 256, 0, stream>>>(kvs, scal, kvsT);
  k_denomv<<<(N_NODES + 255) / 256, 256, 0, stream>>>(q, ksum, scal, denomv);
  k_attn_out<<<GB, 256, 0, stream>>>(q, kvsT, scal, v, h, denomv, t_ln1_g, t_ln1_b, xt);

  // --- hetero GNN branch (reuses h/q/k/v buffers) ---
  k_degree<<<(N_EDGES + 255) / 256, 256, 0, stream>>>(ei, et, deg0, deg1);
  k_gemm_bn_relu<<<GB, 256, 0, stream>>>(x, gfcT, g_fc_b, bn0_g, bn0_b, bn0_m, bn0_v,
                                         nullptr, g);
  k_zero<<<(N_NODES * HID + 255) / 256, 256, 0, stream>>>(agg, N_NODES * HID);
  k_spmm<<<N_EDGES / 8, 256, 0, stream>>>(g, agg, ei, et, ew, deg0, 0);
  k_gemm_bn_relu<<<GB, 256, 0, stream>>>(agg, gw1T, g_b1, bn1_g, bn1_b, bn1_m, bn1_v,
                                         g, g1);
  k_zero<<<(N_NODES * HID + 255) / 256, 256, 0, stream>>>(agg, N_NODES * HID);
  k_spmm<<<N_EDGES / 8, 256, 0, stream>>>(g1, agg, ei, et, ew, deg1, 1);
  k_gemm_bn_relu<<<GB, 256, 0, stream>>>(agg, gw2T, g_b2, bn2_g, bn2_b, bn2_m, bn2_v,
                                         g1, g2);

  // --- output head ---
  k_out<<<GB, 256, 0, stream>>>(xt, g2, outT, out_b, (float*)d_out);

  (void)in_sizes; (void)n_in; (void)out_size; (void)ws_size;
}
